// Mamba2_48636209660191
// MI455X (gfx1250) — matrix operand order repaired and checked by reading, not compiled, not run on hardware
//
#include <hip/hip_runtime.h>
#include <hip/hip_bf16.h>
#include <stdint.h>

#define EMBED    2048
#define D_INNER  4096
#define HEADDIM  128
#define NHEADS   32
#define DSTATE   64
#define DCONV    4
#define CHUNK    256
#define CONV_DIM 4224            // D_INNER + 2*DSTATE
#define DPROJ    8352            // 2*D_INNER + 2*DSTATE + NHEADS
#define BATCH    2
#define SEQ      4096
#define NCHUNKS  16              // SEQ / CHUNK
#define BT       8192            // BATCH*SEQ

typedef unsigned short u16;
typedef unsigned int   u32;
typedef __attribute__((ext_vector_type(16))) __bf16 v16bf;
typedef __attribute__((ext_vector_type(8)))  float  v8f;

struct U4 { u32 x, y, z, w; };
union Frag { u16 s[16]; U4 q[2]; v16bf v; };

__device__ __forceinline__ u16 f2bf(float f) {
    u32 u = __float_as_uint(f);
    u32 r = u + 0x7fffu + ((u >> 16) & 1u);
    return (u16)(r >> 16);
}
__device__ __forceinline__ float bf2f(u16 s) {
    return __uint_as_float(((u32)s) << 16);
}
__device__ __forceinline__ v8f vzero() {
    v8f z = {0.f, 0.f, 0.f, 0.f, 0.f, 0.f, 0.f, 0.f};
    return z;
}
__device__ __forceinline__ v8f wmma_bf16(v16bf a, v16bf b, v8f c) {
    return __builtin_amdgcn_wmma_f32_16x16x32_bf16(false, a, false, b, (short)0, c, false, false);
}

// ---------------------------------------------------------------- converts
__global__ void k_cvt_bf16(const float* __restrict__ src, u16* __restrict__ dst, int n) {
    int i = blockIdx.x * blockDim.x + threadIdx.x;
    if (i < n) dst[i] = f2bf(src[i]);
}

// ---------------------------------------------------------------- bf16 WMMA GEMM: Y[m,n] = sum_k A[m*K+k]*Bw[n*K+k]
// block = 256 thr (8 waves), block tile 128(M) x 128(N), wave tile 32x64
__global__ void k_gemm_bf16(const u16* __restrict__ A, const u16* __restrict__ Bw,
                            float* __restrict__ Y, int N, int K) {
    const int lane = threadIdx.x & 31;
    const int wave = threadIdx.x >> 5;
    const int lm   = lane & 15;
    const int m0 = blockIdx.x * 128 + (wave >> 1) * 32;
    const int n0 = blockIdx.y * 128 + (wave & 1) * 64;
    const int kaoff = (lane >= 16) ? 16 : 0;
    const int kboff = (lane >= 16) ? 16 : 0;

    v8f acc[2][4];
#pragma unroll
    for (int i = 0; i < 2; ++i)
#pragma unroll
        for (int j = 0; j < 4; ++j) acc[i][j] = vzero();

    for (int k0 = 0; k0 < K; k0 += 32) {
        Frag af[2];
#pragma unroll
        for (int i = 0; i < 2; ++i) {
            const u16* p = A + (size_t)(m0 + i * 16 + lm) * K + k0 + kaoff;
            af[i].q[0] = *(const U4*)(p);
            af[i].q[1] = *(const U4*)(p + 8);
            __builtin_prefetch(p + 32, 0, 1);
        }
        Frag bf[4];
#pragma unroll
        for (int j = 0; j < 4; ++j) {
            int n = n0 + j * 16 + lm;
            if (n >= N) n = N - 1;               // clamp (store is guarded)
            const u16* p = Bw + (size_t)n * K + k0 + kboff;
            bf[j].q[0] = *(const U4*)(p);
            bf[j].q[1] = *(const U4*)(p + 8);
        }
#pragma unroll
        for (int i = 0; i < 2; ++i)
#pragma unroll
            for (int j = 0; j < 4; ++j)
                acc[i][j] = wmma_bf16(af[i].v, bf[j].v, acc[i][j]);
    }
#pragma unroll
    for (int i = 0; i < 2; ++i)
#pragma unroll
        for (int j = 0; j < 4; ++j) {
            const int col = n0 + j * 16 + lm;
            if (col < N) {
#pragma unroll
                for (int r = 0; r < 8; ++r) {
                    const int row = m0 + i * 16 + ((lane < 16) ? r : 8 + r);
                    Y[(size_t)row * N + col] = acc[i][j][r];
                }
            }
        }
}

// ---------------------------------------------------------------- depthwise conv + SiLU + split + softplus(dt)
__global__ void k_conv_split(const float* __restrict__ zx, const float* __restrict__ conv_w,
                             const float* __restrict__ conv_b, const float* __restrict__ dt_bias,
                             u16* __restrict__ xbf, u16* __restrict__ Bbf, u16* __restrict__ Cbf,
                             float* __restrict__ dtout) {
    const int ch = blockIdx.x * blockDim.x + threadIdx.x;   // 0..4255
    const int bl = blockIdx.y;                              // 0..BT-1
    const int t  = bl & (SEQ - 1);
    if (ch < CONV_DIM) {
        float acc = conv_b[ch];
#pragma unroll
        for (int d = 0; d < DCONV; ++d) {
            const int tt = t - (DCONV - 1) + d;
            if (tt >= 0)
                acc += zx[(size_t)(bl + d - (DCONV - 1)) * DPROJ + D_INNER + ch] * conv_w[ch * DCONV + d];
        }
        const float y = acc / (1.f + expf(-acc));           // SiLU
        if (ch < D_INNER)            xbf[(size_t)bl * D_INNER + ch] = f2bf(y);
        else if (ch < D_INNER + DSTATE) Bbf[(size_t)bl * DSTATE + (ch - D_INNER)] = f2bf(y);
        else                         Cbf[(size_t)bl * DSTATE + (ch - D_INNER - DSTATE)] = f2bf(y);
    } else if (ch < CONV_DIM + NHEADS) {
        const int h = ch - CONV_DIM;
        const float raw = zx[(size_t)bl * DPROJ + D_INNER + CONV_DIM + h] + dt_bias[h];
        dtout[(size_t)bl * NHEADS + h] = (raw > 20.f) ? raw : log1pf(expf(raw));
    }
}

// ---------------------------------------------------------------- per-(b,c,h) cumsum of dt*A
__global__ void k_cumsum(const float* __restrict__ dt, const float* __restrict__ A_log,
                         float* __restrict__ a, float* __restrict__ a_last) {
    const int i = blockIdx.x * blockDim.x + threadIdx.x;
    if (i >= BATCH * NCHUNKS * NHEADS) return;
    const int h  = i & (NHEADS - 1);
    const int bc = i >> 5;
    const float Ah = -expf(A_log[h]);
    const int l0 = bc * CHUNK;
    float s = 0.f;
    for (int t = 0; t < CHUNK; ++t) {
        s += dt[(size_t)(l0 + t) * NHEADS + h] * Ah;
        a[(size_t)(l0 + t) * NHEADS + h] = s;
    }
    a_last[i] = s;
}

// ---------------------------------------------------------------- scores[b,c,i,j] = sum_n C[i,n]*B[j,n]  (WMMA, K=64)
__global__ void k_scores(const u16* __restrict__ Cbf, const u16* __restrict__ Bbf,
                         float* __restrict__ scores) {
    const int bc = blockIdx.x;
    const int lane = threadIdx.x & 31, wave = threadIdx.x >> 5, lm = lane & 15;
    const int l0 = bc * CHUNK;
    const int kaoff = (lane >= 16) ? 16 : 0;
    const int kboff = (lane >= 16) ? 16 : 0;
    float* sc = scores + (size_t)bc * CHUNK * CHUNK;
    for (int tile = wave; tile < 256; tile += 8) {
        const int ti = tile >> 4, tj = tile & 15;
        v8f acc = vzero();
#pragma unroll
        for (int k0 = 0; k0 < DSTATE; k0 += 32) {
            Frag af, bf;
            const u16* pa = Cbf + (size_t)(l0 + ti * 16 + lm) * DSTATE + k0 + kaoff;
            af.q[0] = *(const U4*)pa;
            af.q[1] = *(const U4*)(pa + 8);
            const u16* pb = Bbf + (size_t)(l0 + tj * 16 + lm) * DSTATE + k0 + kboff;
            bf.q[0] = *(const U4*)pb;
            bf.q[1] = *(const U4*)(pb + 8);
            acc = wmma_bf16(af.v, bf.v, acc);
        }
#pragma unroll
        for (int r = 0; r < 8; ++r) {
            const int i = ti * 16 + ((lane < 16) ? r : 8 + r);
            const int j = tj * 16 + lm;
            sc[i * CHUNK + j] = acc[r];
        }
    }
}

// ---------------------------------------------------------------- per-(b,c,h): y_intra + D*x, and chunk states
__global__ void k_chunk(const u16* __restrict__ xbf, const u16* __restrict__ Bbf,
                        const float* __restrict__ scores, const float* __restrict__ dt,
                        const float* __restrict__ a, const float* __restrict__ a_last,
                        const float* __restrict__ D_skip,
                        float* __restrict__ y, float* __restrict__ states) {
    __shared__ float a_s[CHUNK], dt_s[CHUNK], g_s[CHUNK];
    const int bch = blockIdx.x;
    const int h = bch & (NHEADS - 1), bc = bch >> 5;
    const int l0 = bc * CHUNK;
    const int tid = threadIdx.x, lane = tid & 31, wave = tid >> 5, lm = lane & 15;
    if (tid < CHUNK) {
        const float av = a[(size_t)(l0 + tid) * NHEADS + h];
        const float dv = dt[(size_t)(l0 + tid) * NHEADS + h];
        a_s[tid] = av; dt_s[tid] = dv;
        g_s[tid] = dv * expf(a_last[bch] - av);
    }
    __syncthreads();
    const float* sc = scores + (size_t)bc * CHUNK * CHUNK;
    const int kaoff = (lane >= 16) ? 16 : 0;
    const int kboff = (lane >= 16) ? 16 : 0;

    // ---- Phase A: y_intra = W(256x256) @ x(256x128), W built on the fly
    {
        const int m0 = wave * 32;
        v8f acc[2][8];
#pragma unroll
        for (int i = 0; i < 2; ++i)
#pragma unroll
            for (int j = 0; j < 8; ++j) acc[i][j] = vzero();

        for (int ks = 0; ks < 8; ++ks) {
            const int j0 = ks * 32;
            Frag af[2];
#pragma unroll
            for (int mi = 0; mi < 2; ++mi) {
                const int i = m0 + mi * 16 + lm;
                const float ai = a_s[i];
#pragma unroll
                for (int e = 0; e < 16; ++e) {
                    const int j = j0 + kaoff + e;
                    float wv = 0.f;
                    if (i >= j) wv = sc[i * CHUNK + j] * expf(ai - a_s[j]) * dt_s[j];
                    af[mi].s[e] = f2bf(wv);
                }
            }
#pragma unroll
            for (int nt = 0; nt < 8; ++nt) {
                Frag bf;
                const int p = nt * 16 + lm;
#pragma unroll
                for (int e = 0; e < 16; ++e) {
                    const int t = j0 + kboff + e;
                    bf.s[e] = xbf[(size_t)(l0 + t) * D_INNER + h * HEADDIM + p];
                }
                acc[0][nt] = wmma_bf16(af[0].v, bf.v, acc[0][nt]);
                acc[1][nt] = wmma_bf16(af[1].v, bf.v, acc[1][nt]);
            }
        }
        const float Dh = D_skip[h];
#pragma unroll
        for (int mi = 0; mi < 2; ++mi)
#pragma unroll
            for (int nt = 0; nt < 8; ++nt) {
                const int p = nt * 16 + lm;
#pragma unroll
                for (int r = 0; r < 8; ++r) {
                    const int i = m0 + mi * 16 + ((lane < 16) ? r : 8 + r);
                    const size_t idx = (size_t)(l0 + i) * D_INNER + h * HEADDIM + p;
                    y[idx] = acc[mi][nt][r] + Dh * bf2f(xbf[idx]);
                }
            }
    }

    // ---- Phase B: states(128x64) = x^T(128x256) @ (B*g)(256x64)
    {
        const int p0 = wave * 16;
        v8f sacc[4];
#pragma unroll
        for (int j = 0; j < 4; ++j) sacc[j] = vzero();
        for (int ks = 0; ks < 8; ++ks) {
            const int t0 = ks * 32;
            Frag af;
            const int p = p0 + lm;
#pragma unroll
            for (int e = 0; e < 16; ++e) {
                const int t = t0 + kaoff + e;
                af.s[e] = xbf[(size_t)(l0 + t) * D_INNER + h * HEADDIM + p];
            }
#pragma unroll
            for (int nt = 0; nt < 4; ++nt) {
                Frag bf;
                const int n = nt * 16 + lm;
#pragma unroll
                for (int e = 0; e < 16; ++e) {
                    const int t = t0 + kboff + e;
                    bf.s[e] = f2bf(bf2f(Bbf[(size_t)(l0 + t) * DSTATE + n]) * g_s[t]);
                }
                sacc[nt] = wmma_bf16(af.v, bf.v, sacc[nt]);
            }
        }
#pragma unroll
        for (int nt = 0; nt < 4; ++nt) {
            const int n = nt * 16 + lm;
#pragma unroll
            for (int r = 0; r < 8; ++r) {
                const int p = p0 + ((lane < 16) ? r : 8 + r);
                states[((size_t)bch * HEADDIM + p) * DSTATE + n] = sacc[nt][r];
            }
        }
    }
}

// ---------------------------------------------------------------- inter-chunk sequential scan
__global__ void k_scan(const float* __restrict__ states, const float* __restrict__ a_last,
                       float* __restrict__ prev) {
    const int i = blockIdx.x * blockDim.x + threadIdx.x;
    if (i >= BATCH * NHEADS * HEADDIM * DSTATE) return;
    const int n = i & 63;
    const int p = (i >> 6) & 127;
    const int h = (i >> 13) & 31;
    const int b = i >> 18;
    float s = 0.f;
    for (int c = 0; c < NCHUNKS; ++c) {
        const int bch = (b * NCHUNKS + c) * NHEADS + h;
        const size_t off = ((size_t)bch * HEADDIM + p) * DSTATE + n;
        prev[off] = s;
        s = s * expf(a_last[bch]) + states[off];
    }
}

// ---------------------------------------------------------------- y += exp(a_i) * (C @ prev^T)
__global__ void k_inter(const u16* __restrict__ Cbf, const float* __restrict__ prev,
                        const float* __restrict__ a, float* __restrict__ y) {
    __shared__ u16 pv[HEADDIM * DSTATE];
    __shared__ float ea_s[CHUNK];
    const int bch = blockIdx.x;
    const int h = bch & (NHEADS - 1), bc = bch >> 5;
    const int l0 = bc * CHUNK;
    const int tid = threadIdx.x, lane = tid & 31, wave = tid >> 5, lm = lane & 15;
    for (int i = tid; i < HEADDIM * DSTATE; i += 256)
        pv[i] = f2bf(prev[(size_t)bch * HEADDIM * DSTATE + i]);
    if (tid < CHUNK) ea_s[tid] = expf(a[(size_t)(l0 + tid) * NHEADS + h]);
    __syncthreads();

    const int m0 = wave * 32;
    const int kaoff = (lane >= 16) ? 16 : 0;
    const int kboff = (lane >= 16) ? 16 : 0;
    v8f acc[2][8];
#pragma unroll
    for (int i = 0; i < 2; ++i)
#pragma unroll
        for (int j = 0; j < 8; ++j) acc[i][j] = vzero();

#pragma unroll
    for (int ks = 0; ks < 2; ++ks) {
        const int k0 = ks * 32;
        Frag af[2];
#pragma unroll
        for (int mi = 0; mi < 2; ++mi) {
            const u16* pa = Cbf + (size_t)(l0 + m0 + mi * 16 + lm) * DSTATE + k0 + kaoff;
            af[mi].q[0] = *(const U4*)pa;
            af[mi].q[1] = *(const U4*)(pa + 8);
        }
#pragma unroll
        for (int nt = 0; nt < 8; ++nt) {
            Frag bf;
            const int p = nt * 16 + lm;
#pragma unroll
            for (int e = 0; e < 16; ++e) {
                const int n = k0 + kboff + e;
                bf.s[e] = pv[p * DSTATE + n];
            }
            acc[0][nt] = wmma_bf16(af[0].v, bf.v, acc[0][nt]);
            acc[1][nt] = wmma_bf16(af[1].v, bf.v, acc[1][nt]);
        }
    }
#pragma unroll
    for (int mi = 0; mi < 2; ++mi)
#pragma unroll
        for (int nt = 0; nt < 8; ++nt) {
            const int p = nt * 16 + lm;
#pragma unroll
            for (int r = 0; r < 8; ++r) {
                const int i = m0 + mi * 16 + ((lane < 16) ? r : 8 + r);
                const size_t idx = (size_t)(l0 + i) * D_INNER + h * HEADDIM + p;
                y[idx] += acc[mi][nt][r] * ea_s[i];
            }
        }
}

// ---------------------------------------------------------------- gating + per-head RMS norm -> bf16
__global__ void k_gate_norm(const float* __restrict__ y, const float* __restrict__ zx,
                            const float* __restrict__ norm_w, u16* __restrict__ ynbf) {
    const int row = blockIdx.x * 8 + (threadIdx.x >> 5);  // (bl,h) rows
    const int lane = threadIdx.x & 31;
    const int h = row & (NHEADS - 1);
    const int bl = row >> 5;
    float v[4];
    float ss = 0.f;
#pragma unroll
    for (int k = 0; k < 4; ++k) {
        const int d = h * HEADDIM + lane + 32 * k;
        const float yy = y[(size_t)bl * D_INNER + d];
        const float zz = zx[(size_t)bl * DPROJ + d];
        const float g = yy * (1.f / (1.f + expf(-zz)));
        v[k] = g;
        ss += g * g;
    }
#pragma unroll
    for (int off = 16; off > 0; off >>= 1) ss += __shfl_xor(ss, off, 32);
    const float rinv = 1.f / sqrtf(ss * (1.f / HEADDIM) + 1e-6f);
#pragma unroll
    for (int k = 0; k < 4; ++k) {
        const int d = h * HEADDIM + lane + 32 * k;
        ynbf[(size_t)bl * D_INNER + d] = f2bf(v[k] * rinv * norm_w[d]);
    }
}

// ================================================================ launcher
extern "C" void kernel_launch(void* const* d_in, const int* in_sizes, int n_in,
                              void* d_out, int out_size, void* d_ws, size_t ws_size,
                              hipStream_t stream) {
    const float* u         = (const float*)d_in[0];
    const float* in_proj_w = (const float*)d_in[1];
    const float* conv_w    = (const float*)d_in[2];
    const float* conv_b    = (const float*)d_in[3];
    const float* A_log     = (const float*)d_in[4];
    const float* dt_bias   = (const float*)d_in[5];
    const float* D_skip    = (const float*)d_in[6];
    const float* norm_w    = (const float*)d_in[7];
    const float* o_proj_w  = (const float*)d_in[8];
    float* out = (float*)d_out;

    char* ws = (char*)d_ws;
    size_t off = 0;
    auto take = [&](size_t bytes) -> char* {
        char* p = ws + off;
        off += (bytes + 255) & ~(size_t)255;
        return p;
    };
    u16*   ubf    = (u16*)  take((size_t)BT * EMBED * 2);
    u16*   wbf    = (u16*)  take((size_t)DPROJ * EMBED * 2);
    u16*   obf    = (u16*)  take((size_t)EMBED * D_INNER * 2);
    float* zx     = (float*)take((size_t)BT * DPROJ * 4);
    u16*   xbf    = (u16*)  take((size_t)BT * D_INNER * 2);
    u16*   Bbf    = (u16*)  take((size_t)BT * DSTATE * 2);
    u16*   Cbf    = (u16*)  take((size_t)BT * DSTATE * 2);
    float* dtb    = (float*)take((size_t)BT * NHEADS * 4);
    float* av     = (float*)take((size_t)BT * NHEADS * 4);
    float* alast  = (float*)take((size_t)BATCH * NCHUNKS * NHEADS * 4);
    float* scores = (float*)take((size_t)BATCH * NCHUNKS * CHUNK * CHUNK * 4);
    float* states = (float*)take((size_t)BATCH * NCHUNKS * NHEADS * HEADDIM * DSTATE * 4);
    float* prev   = (float*)take((size_t)BATCH * NCHUNKS * NHEADS * HEADDIM * DSTATE * 4);
    float* yv     = (float*)take((size_t)BT * D_INNER * 4);
    u16*   ynbf   = (u16*)  take((size_t)BT * D_INNER * 2);

    // 1) bf16 conversions (weights + activations become L2-resident working set)
    k_cvt_bf16<<<(BT * EMBED) / 256, 256, 0, stream>>>(u, ubf, BT * EMBED);
    k_cvt_bf16<<<(DPROJ * EMBED) / 256, 256, 0, stream>>>(in_proj_w, wbf, DPROJ * EMBED);
    k_cvt_bf16<<<(EMBED * D_INNER) / 256, 256, 0, stream>>>(o_proj_w, obf, EMBED * D_INNER);

    // 2) in_proj GEMM: zx = u @ in_proj_w^T  (8192 x 8352, K=2048)
    k_gemm_bf16<<<dim3(BT / 128, (DPROJ + 127) / 128), 256, 0, stream>>>(ubf, wbf, zx, DPROJ, EMBED);

    // 3) depthwise conv + SiLU + split + softplus(dt)
    k_conv_split<<<dim3((CONV_DIM + NHEADS + 255) / 256, BT), 256, 0, stream>>>(
        zx, conv_w, conv_b, dt_bias, xbf, Bbf, Cbf, dtb);

    // 4) cumsum of dt*A per (b,c,h)
    k_cumsum<<<(BATCH * NCHUNKS * NHEADS + 255) / 256, 256, 0, stream>>>(dtb, A_log, av, alast);

    // 5) chunk scores (shared across heads, NGROUPS=1)
    k_scores<<<BATCH * NCHUNKS, 256, 0, stream>>>(Cbf, Bbf, scores);

    // 6) intra-chunk output + chunk states
    k_chunk<<<BATCH * NCHUNKS * NHEADS, 256, 0, stream>>>(
        xbf, Bbf, scores, dtb, av, alast, D_skip, yv, states);

    // 7) inter-chunk state scan
    k_scan<<<(BATCH * NHEADS * HEADDIM * DSTATE) / 256, 256, 0, stream>>>(states, alast, prev);

    // 8) inter-chunk contribution
    k_inter<<<BATCH * NCHUNKS * NHEADS, 256, 0, stream>>>(Cbf, prev, av, yv);

    // 9) gating + RMS norm
    k_gate_norm<<<(BT * NHEADS) / 8, 256, 0, stream>>>(yv, zx, norm_w, ynbf);

    // 10) out_proj GEMM: out = yn @ o_proj_w^T  (8192 x 2048, K=4096)
    k_gemm_bf16<<<dim3(BT / 128, EMBED / 128), 256, 0, stream>>>(ynbf, obf, out, EMBED, D_INNER);
}
